// SFBlock_81106162418333
// MI455X (gfx1250) — matrix operand order repaired and checked by reading, compiled, not run on hardware
//
#include <hip/hip_runtime.h>
#include <hip/hip_bf16.h>
#include <math.h>

typedef __attribute__((ext_vector_type(16))) _Float16 v16h;
typedef __attribute__((ext_vector_type(8)))  _Float16 v8h;
typedef __attribute__((ext_vector_type(8)))  float    v8f;
typedef __attribute__((ext_vector_type(4)))  unsigned int u32x4;
typedef __attribute__((ext_vector_type(8)))  int      i32x8;
typedef __attribute__((ext_vector_type(4)))  int      i32x4;

#define CC    256
#define LL    4096           // H*W
#define BB    16
#define HEADS 8
#define HC    32
#define FH    512            // FFN_HIDDEN

#if defined(__has_builtin)
#  if __has_builtin(__builtin_amdgcn_tensor_load_to_lds) && __has_builtin(__builtin_amdgcn_s_wait_tensorcnt)
#    define HAVE_TDM 1
#  else
#    define HAVE_TDM 0
#  endif
#else
#  define HAVE_TDM 0
#endif

__device__ __forceinline__ v16h combine16(v8h lo, v8h hi) {
    v16h r;
#pragma unroll
    for (int i = 0; i < 8; ++i) { r[i] = lo[i]; r[i + 8] = hi[i]; }
    return r;
}

// A-fragment (16x32 f16): lane l -> row (l&15); lanes<16 hold K k0+0..7,k0+16..23; lanes>=16 k0+8..15,k0+24..31
template <int KP>
__device__ __forceinline__ v16h load_a_frag(const _Float16* As, int row0, int k0, int lane) {
    int row = row0 + (lane & 15);
    int kb  = k0 + (lane >> 4) * 8;
    v8h lo = *(const v8h*)(As + row * KP + kb);
    v8h hi = *(const v8h*)(As + row * KP + kb + 16);
    return combine16(lo, hi);
}

// B-fragment (32x16 f16), LDS transposed Bs[col][k] stride 40: lane l -> col (l&15); lanes<16 hold K 0..7,16..23; lanes>=16 8..15,24..31
__device__ __forceinline__ v16h load_b_frag(const _Float16* Bs, int col0, int lane) {
    int col = col0 + (lane & 15);
    int kb  = (lane >> 4) * 8;
    v8h lo = *(const v8h*)(Bs + col * 40 + kb);
    v8h hi = *(const v8h*)(Bs + col * 40 + kb + 16);
    return combine16(lo, hi);
}

// ---- TDM: 1-D contiguous panel copy global->LDS with hardware row padding ----
// ELEMS f16 elements; after every (1<<PAD_ICODE) 8-byte units stored, insert 4 dwords (16B = 8 f16) pad.
template <int ELEMS, int PAD_ICODE>
__device__ __forceinline__ void tdm_panel_load(unsigned lds_byte_addr, const _Float16* gsrc) {
#if HAVE_TDM
    unsigned long long ga = (unsigned long long)gsrc;
    u32x4 g0;
    g0[0] = 1u;                                               // count=1, user mode
    g0[1] = lds_byte_addr;                                    // lds_addr
    g0[2] = (unsigned)ga;                                     // global_addr[31:0]
    g0[3] = (unsigned)((ga >> 32) & 0x01FFFFFFull) | (2u << 30); // global_addr[56:32] | type=2
    i32x8 g1;
    g1[0] = (int)((1u << 16) | (1u << 20) | ((unsigned)PAD_ICODE << 22) | (3u << 25));
    //        data_size=2B    pad_enable     pad_interval           pad_amount=4 dwords
    g1[1] = (int)(((unsigned)ELEMS & 0xFFFFu) << 16);         // tensor_dim0[15:0] @ bits[63:48]
    g1[2] = (int)((((unsigned)ELEMS >> 16) & 0xFFFFu) | (1u << 16)); // tensor_dim0[31:16] | tensor_dim1=1
    g1[3] = (int)(((unsigned)ELEMS & 0xFFFFu) << 16);         // tile_dim0 @ bits[127:112]
    g1[4] = 0;                                                // tile_dim1=0, tile_dim2=0 (unused)
    g1[5] = (int)ELEMS;                                       // tensor_dim0_stride[31:0]
    g1[6] = 0;
    g1[7] = 0;
    i32x4 z4 = {0, 0, 0, 0};
#if __clang_major__ >= 23
    i32x8 z8 = {0, 0, 0, 0, 0, 0, 0, 0};
    __builtin_amdgcn_tensor_load_to_lds(g0, g1, z4, z4, z8, 0);
#else
    __builtin_amdgcn_tensor_load_to_lds(g0, g1, z4, z4, 0);
#endif
#endif
}

// Fallback cooperative panel copy (f16 -> padded-row LDS), 256 threads
template <int K>
__device__ __forceinline__ void stage_panel(const _Float16* g, _Float16* As, int tid) {
    constexpr int KP = K + 8;
#pragma unroll
    for (int i = tid * 8; i < 64 * K; i += 256 * 8) {
        int r = i / K, c = i % K;
        *(v8h*)(As + r * KP + c) = *(const v8h*)(g + i);
    }
}

// -------- LayerNorm over channels (biased var, eps=1e-5), f32 in -> f16 out, layout (B,C,L) --------
__global__ __launch_bounds__(256) void ln_kernel(const float* __restrict__ X,
                                                 const float* __restrict__ g,
                                                 const float* __restrict__ be,
                                                 _Float16* __restrict__ Y) {
    long p = (long)blockIdx.x * 256 + threadIdx.x;
    int b = (int)(p >> 12);
    int l = (int)(p & (LL - 1));
    const float* xp = X + (long)b * CC * LL + l;
    float s = 0.f, sq = 0.f;
#pragma unroll 8
    for (int c = 0; c < CC; ++c) { float v = xp[(long)c * LL]; s += v; sq += v * v; }
    float mu  = s * (1.0f / CC);
    float var = sq * (1.0f / CC) - mu * mu;
    float rs  = rsqrtf(var + 1e-5f);
    _Float16* yp = Y + (long)b * CC * LL + l;
#pragma unroll 8
    for (int c = 0; c < CC; ++c)
        yp[(long)c * LL] = (_Float16)((xp[(long)c * LL] - mu) * rs * g[c] + be[c]);
}

// -------- f32 -> f16 weight conversion --------
__global__ __launch_bounds__(256) void cvt_f16(const float* __restrict__ src,
                                               _Float16* __restrict__ dst, int n) {
    int i = blockIdx.x * 256 + threadIdx.x;
    if (i < n) dst[i] = (_Float16)src[i];
}

// -------- WMMA GEMM: Y[b] = epilogue( A[b?] @ X[b] ); A (M,K) f16, X (B,K,L) f16 --------
// EPI: 0 -> Yh f16 = acc ; 1 -> Yf f32 = Res + acc ; 2 -> Yf f32 = Res + acc + bias[m]
template <int K, int EPI>
__global__ __launch_bounds__(256) void gemm_wmma(const _Float16* __restrict__ Ah,
                                                 const _Float16* __restrict__ X,
                                                 _Float16* __restrict__ Yh,
                                                 float* __restrict__ Yf,
                                                 const float* __restrict__ Res,
                                                 const float* __restrict__ bias,
                                                 int M, long aBatchStride) {
    constexpr int KP = K + 8;                       // padded LDS row stride (conflict-free, 16B aligned)
    __shared__ _Float16 As[64 * KP];                // full-K weight panel, staged once
    __shared__ _Float16 Bs[128 * 40];               // transposed activation tile
    const int t  = threadIdx.x;
    const int b  = blockIdx.z;
    const int m0 = blockIdx.y * 64;
    const int n0 = blockIdx.x * 128;
    const int wave = t >> 5, lane = t & 31;
    const int wm = wave >> 2, wn = wave & 3;
    const long xbase = (long)b * K * LL;
    const _Float16* Apanel = Ah + (long)b * aBatchStride + (long)m0 * K;

    // ---- stage the whole 64xK A panel via the Tensor Data Mover (one DMA, padded rows) ----
#if HAVE_TDM
    if (t < 32)
        tdm_panel_load<64 * K, (K == 256 ? 6 : 7)>((unsigned)(unsigned long long)(void*)As, Apanel);
#else
    stage_panel<K>(Apanel, As, t);
#endif

    v8f acc[2][2] = {};

    for (int k0 = 0; k0 < K; k0 += 32) {
        // ---- stage B tile (32 x 128) transposed into Bs[col][k] ----
        {
            int kr0 = t >> 4;
            int c8  = (t & 15) * 8;
#pragma unroll
            for (int i = 0; i < 2; ++i) {
                int kr = kr0 + i * 16;
                v8h vv = *(const v8h*)(X + xbase + (long)(k0 + kr) * LL + n0 + c8);
#pragma unroll
                for (int j = 0; j < 8; ++j) Bs[(c8 + j) * 40 + kr] = vv[j];
            }
        }
#if HAVE_TDM
        if (k0 == 0 && t < 32) __builtin_amdgcn_s_wait_tensorcnt(0);
#endif
        __syncthreads();

        v16h afr[2], bfr[2];
#pragma unroll
        for (int mi = 0; mi < 2; ++mi) afr[mi] = load_a_frag<KP>(As, wm * 32 + mi * 16, k0, lane);
#pragma unroll
        for (int ni = 0; ni < 2; ++ni) bfr[ni] = load_b_frag(Bs, wn * 32 + ni * 16, lane);
#pragma unroll
        for (int mi = 0; mi < 2; ++mi)
#pragma unroll
            for (int ni = 0; ni < 2; ++ni)
                acc[mi][ni] = __builtin_amdgcn_wmma_f32_16x16x32_f16(
                    false, afr[mi], false, bfr[ni], (short)0, acc[mi][ni], false, false);
        __syncthreads();
    }

    int colL = lane & 15, rb = (lane >> 4) * 8;
#pragma unroll
    for (int mi = 0; mi < 2; ++mi)
#pragma unroll
        for (int ni = 0; ni < 2; ++ni) {
            int m = m0 + wm * 32 + mi * 16 + rb;
            int n = n0 + wn * 32 + ni * 16 + colL;
            long base = ((long)b * M + m) * LL + n;
#pragma unroll
            for (int r = 0; r < 8; ++r) {
                long idx = base + (long)r * LL;
                float v  = acc[mi][ni][r];
                if constexpr (EPI == 0)      Yh[idx] = (_Float16)v;
                else if constexpr (EPI == 1) Yf[idx] = Res[idx] + v;
                else                         Yf[idx] = Res[idx] + v + bias[m + r];
            }
        }
}

// -------- Gated FFN GEMM: G[j] = (W1[j]@X + b1[j]) * (W1[j+512]@X + b1[j+512]), j in [0,512) --------
__global__ __launch_bounds__(256) void gemm_gate(const _Float16* __restrict__ W1h,
                                                 const float* __restrict__ b1,
                                                 const _Float16* __restrict__ X,
                                                 _Float16* __restrict__ G) {
    constexpr int KP = CC + 8;
    __shared__ _Float16 As1[64 * KP];
    __shared__ _Float16 As2[64 * KP];
    __shared__ _Float16 Bs[128 * 40];
    const int t  = threadIdx.x;
    const int b  = blockIdx.z;
    const int m0 = blockIdx.y * 64;
    const int n0 = blockIdx.x * 128;
    const int wave = t >> 5, lane = t & 31;
    const int wm = wave >> 2, wn = wave & 3;
    const long xbase = (long)b * CC * LL;

#if HAVE_TDM
    if (t < 32) {
        tdm_panel_load<64 * CC, 6>((unsigned)(unsigned long long)(void*)As1, W1h + (long)m0 * CC);
        tdm_panel_load<64 * CC, 6>((unsigned)(unsigned long long)(void*)As2, W1h + (long)(m0 + FH) * CC);
    }
#else
    stage_panel<CC>(W1h + (long)m0 * CC, As1, t);
    stage_panel<CC>(W1h + (long)(m0 + FH) * CC, As2, t);
#endif

    v8f acc1[2][2] = {};
    v8f acc2[2][2] = {};

    for (int k0 = 0; k0 < CC; k0 += 32) {
        {
            int kr0 = t >> 4;
            int c8  = (t & 15) * 8;
#pragma unroll
            for (int i = 0; i < 2; ++i) {
                int kr = kr0 + i * 16;
                v8h vv = *(const v8h*)(X + xbase + (long)(k0 + kr) * LL + n0 + c8);
#pragma unroll
                for (int j = 0; j < 8; ++j) Bs[(c8 + j) * 40 + kr] = vv[j];
            }
        }
#if HAVE_TDM
        if (k0 == 0 && t < 32) __builtin_amdgcn_s_wait_tensorcnt(0);
#endif
        __syncthreads();

        v16h a1[2], a2[2], bfr[2];
#pragma unroll
        for (int mi = 0; mi < 2; ++mi) {
            a1[mi] = load_a_frag<KP>(As1, wm * 32 + mi * 16, k0, lane);
            a2[mi] = load_a_frag<KP>(As2, wm * 32 + mi * 16, k0, lane);
        }
#pragma unroll
        for (int ni = 0; ni < 2; ++ni) bfr[ni] = load_b_frag(Bs, wn * 32 + ni * 16, lane);
#pragma unroll
        for (int mi = 0; mi < 2; ++mi)
#pragma unroll
            for (int ni = 0; ni < 2; ++ni) {
                acc1[mi][ni] = __builtin_amdgcn_wmma_f32_16x16x32_f16(
                    false, a1[mi], false, bfr[ni], (short)0, acc1[mi][ni], false, false);
                acc2[mi][ni] = __builtin_amdgcn_wmma_f32_16x16x32_f16(
                    false, a2[mi], false, bfr[ni], (short)0, acc2[mi][ni], false, false);
            }
        __syncthreads();
    }

    int colL = lane & 15, rb = (lane >> 4) * 8;
#pragma unroll
    for (int mi = 0; mi < 2; ++mi)
#pragma unroll
        for (int ni = 0; ni < 2; ++ni) {
            int m = m0 + wm * 32 + mi * 16 + rb;
            int n = n0 + wn * 32 + ni * 16 + colL;
#pragma unroll
            for (int r = 0; r < 8; ++r) {
                float h1 = acc1[mi][ni][r] + b1[m + r];
                float h2 = acc2[mi][ni][r] + b1[m + r + FH];
                G[((long)b * FH + m + r) * LL + n] = (_Float16)(h1 * h2);
            }
        }
}

// -------- Attention scores + softmax: per (b,h): A = softmax(scale * q_h @ k_h^T) (32x32) --------
__global__ __launch_bounds__(128) void attn_scores(const _Float16* __restrict__ Q,
                                                   const _Float16* __restrict__ Km,
                                                   _Float16* __restrict__ A) {
    __shared__ float Sld[32 * 33];
    const int h = blockIdx.x, b = blockIdx.y;
    const int t = threadIdx.x, wave = t >> 5, lane = t & 31;
    const int mi = wave >> 1, ni = wave & 1;
    const _Float16* qh = Q  + ((long)b * CC + h * HC) * LL;
    const _Float16* kh = Km + ((long)b * CC + h * HC) * LL;

    v8f acc = {};
    const int arow = mi * 16 + (lane & 15);
    const int akb  = (lane >> 4) * 8;
    const int bcol = ni * 16 + (lane & 15);
    const int bkb  = (lane >> 4) * 8;
    for (int k0 = 0; k0 < LL; k0 += 32) {
        v8h alo = *(const v8h*)(qh + (long)arow * LL + k0 + akb);
        v8h ahi = *(const v8h*)(qh + (long)arow * LL + k0 + akb + 16);
        v8h blo = *(const v8h*)(kh + (long)bcol * LL + k0 + bkb);
        v8h bhi = *(const v8h*)(kh + (long)bcol * LL + k0 + bkb + 16);
        acc = __builtin_amdgcn_wmma_f32_16x16x32_f16(
            false, combine16(alo, ahi), false, combine16(blo, bhi), (short)0, acc, false, false);
    }
    int colL = lane & 15, rb = (lane >> 4) * 8;
#pragma unroll
    for (int r = 0; r < 8; ++r)
        Sld[(mi * 16 + rb + r) * 33 + ni * 16 + colL] = acc[r];
    __syncthreads();

    if (t < 32) {
        const float scale = 0.1767766952966369f;   // 1/sqrt(32)
        float mx = -1e30f;
        for (int d = 0; d < 32; ++d) mx = fmaxf(mx, Sld[t * 33 + d] * scale);
        float sum = 0.f;
        for (int d = 0; d < 32; ++d) {
            float e = expf(Sld[t * 33 + d] * scale - mx);
            Sld[t * 33 + d] = e;
            sum += e;
        }
        float inv = 1.0f / sum;
        _Float16* ap = A + (((long)b * HEADS + h) * HC + t) * HC;
        for (int d = 0; d < 32; ++d) ap[d] = (_Float16)(Sld[t * 33 + d] * inv);
    }
}

// -------- Fold: M2[b] = Wo @ blockdiag(A[b])  (256x256 per batch, tiny) --------
__global__ __launch_bounds__(256) void fuse_wo(const float* __restrict__ Wo,
                                               const _Float16* __restrict__ A,
                                               _Float16* __restrict__ M2) {
    const int o = blockIdx.x, b = blockIdx.y, dg = threadIdx.x;
    const int h = dg >> 5, d = dg & 31;
    const float* wrow = Wo + (long)o * CC + h * HC;
    const _Float16* ab = A + (((long)b * HEADS + h) * HC) * HC + d;
    float s = 0.f;
#pragma unroll
    for (int c = 0; c < HC; ++c) s += wrow[c] * (float)ab[c * HC];
    M2[((long)b * CC + o) * CC + dg] = (_Float16)s;
}

extern "C" void kernel_launch(void* const* d_in, const int* in_sizes, int n_in,
                              void* d_out, int out_size, void* d_ws, size_t ws_size,
                              hipStream_t stream) {
    const float* optical = (const float*)d_in[0];
    const float* sar     = (const float*)d_in[1];
    const float* g_opt   = (const float*)d_in[2];
    const float* b_opt   = (const float*)d_in[3];
    const float* g_sar   = (const float*)d_in[4];
    const float* b_sar   = (const float*)d_in[5];
    const float* Wq      = (const float*)d_in[6];
    const float* Wk      = (const float*)d_in[7];
    const float* Wv      = (const float*)d_in[8];
    const float* Wo      = (const float*)d_in[9];
    const float* g_ffn   = (const float*)d_in[10];
    const float* b_ffn   = (const float*)d_in[11];
    const float* W1      = (const float*)d_in[12];
    const float* b1      = (const float*)d_in[13];
    const float* W2      = (const float*)d_in[14];
    const float* b2      = (const float*)d_in[15];
    float* out = (float*)d_out;

    char* ws = (char*)d_ws;
    const size_t SZH = (size_t)BB * CC * LL * sizeof(_Float16);  // 32 MB
    _Float16* P0 = (_Float16*)(ws);                 // opt_n
    _Float16* P1 = (_Float16*)(ws + SZH);           // sar_n
    _Float16* Qb = (_Float16*)(ws + 2 * SZH);       // q  -> later low half of g
    _Float16* Kb = (_Float16*)(ws + 3 * SZH);       // k  -> high half of g
    _Float16* Vb = (_Float16*)(ws + 4 * SZH);       // v  -> later xn
    float*    Xf = (float*)(ws);                    // x f32, aliases P0+P1 (dead by then)
    _Float16* Gf = Qb;                              // gated activations (B,512,L)
    _Float16* XN = Vb;                              // LN(x) f16
    _Float16* Ab  = (_Float16*)(ws + 5 * SZH);                   // softmax A: 256 KB
    _Float16* M2  = (_Float16*)(ws + 5 * SZH + (1u << 20));      // folded Wo@A: 2 MB
    _Float16* WH  = (_Float16*)(ws + 5 * SZH + 3 * (1u << 20));  // f16 weights
    _Float16* Wqh = WH;
    _Float16* Wkh = WH + 65536;
    _Float16* Wvh = WH + 131072;
    _Float16* W2h = WH + 196608;      // 256x512
    _Float16* W1h = WH + 327680;      // 1024x256

    dim3 blk256(256), blk128(128);
    dim3 lnGrid((BB * LL) / 256);
    dim3 gemmGrid(LL / 128, CC / 64, BB);
    dim3 gateGrid(LL / 128, FH / 64, BB);

    // 0. pre-convert all weights to f16 (once)
    cvt_f16<<<256,  blk256, 0, stream>>>(Wq, Wqh, 65536);
    cvt_f16<<<256,  blk256, 0, stream>>>(Wk, Wkh, 65536);
    cvt_f16<<<256,  blk256, 0, stream>>>(Wv, Wvh, 65536);
    cvt_f16<<<512,  blk256, 0, stream>>>(W2, W2h, 131072);
    cvt_f16<<<1024, blk256, 0, stream>>>(W1, W1h, 262144);
    // 1. LayerNorms
    ln_kernel<<<lnGrid, blk256, 0, stream>>>(optical, g_opt, b_opt, P0);
    ln_kernel<<<lnGrid, blk256, 0, stream>>>(sar, g_sar, b_sar, P1);
    // 2. q/k/v projections (TDM-staged weight panels)
    gemm_wmma<CC, 0><<<gemmGrid, blk256, 0, stream>>>(Wqh, P0, Qb, nullptr, nullptr, nullptr, CC, 0);
    gemm_wmma<CC, 0><<<gemmGrid, blk256, 0, stream>>>(Wkh, P1, Kb, nullptr, nullptr, nullptr, CC, 0);
    gemm_wmma<CC, 0><<<gemmGrid, blk256, 0, stream>>>(Wvh, P1, Vb, nullptr, nullptr, nullptr, CC, 0);
    // 3. attention scores + softmax
    attn_scores<<<dim3(HEADS, BB), blk128, 0, stream>>>(Qb, Kb, Ab);
    // 4. fold Wo through block-diagonal A (kills a whole 8.6-GFLOP GEMM pass)
    fuse_wo<<<dim3(CC, BB), blk256, 0, stream>>>(Wo, Ab, M2);
    // 5. x = optical + M2[b] @ v   (f32 out, batched A)
    gemm_wmma<CC, 1><<<gemmGrid, blk256, 0, stream>>>(M2, Vb, nullptr, Xf, optical, nullptr, CC, (long)CC * CC);
    // 6. LN(x)
    ln_kernel<<<lnGrid, blk256, 0, stream>>>(Xf, g_ffn, b_ffn, XN);
    // 7. gated FFN hidden
    gemm_gate<<<gateGrid, blk256, 0, stream>>>(W1h, b1, XN, Gf);
    // 8. out = x + W2 @ g + b2
    gemm_wmma<FH, 2><<<gemmGrid, blk256, 0, stream>>>(W2h, Gf, nullptr, out, Xf, b2, CC, 0);
}
